// Magnus4th_33758442947046
// MI455X (gfx1250) — matrix operand order repaired and checked by reading, not compiled, not run on hardware
//
#include <hip/hip_runtime.h>

#define DD 128
#define BB 1024

typedef __bf16 bf16;
typedef __attribute__((ext_vector_type(16))) __bf16 v16bf;
typedef __attribute__((ext_vector_type(8)))  __bf16 bf16x8;
typedef __attribute__((ext_vector_type(4)))  __bf16 bf16x4;
typedef __attribute__((ext_vector_type(8)))  float  v8f;
typedef __attribute__((ext_vector_type(4)))  float  f32x4;

// float -> bf16 (round to nearest even), bit-exact and guaranteed to compile
__device__ __forceinline__ bf16 f2bf(float x) {
  unsigned u = __builtin_bit_cast(unsigned, x);
  unsigned short s = (unsigned short)((u + 0x7FFFu + ((u >> 16) & 1u)) >> 16);
  return __builtin_bit_cast(bf16, s);
}
__device__ __forceinline__ float bf2f(bf16 v) {
  unsigned short s = __builtin_bit_cast(unsigned short, v);
  unsigned u = ((unsigned)s) << 16;
  return __builtin_bit_cast(float, u);
}

// A-operand (16x32 bf16, MxK) from row-major LDS matrix.
// Contiguous k order (this code's choice): lane<16: M=lane, elems=K(kb+0..15)
//             lane>=16: M=lane-16, elems=K(kb+16..31)
__device__ __forceinline__ v16bf load_A(const bf16* N, int row0, int kb, int lane) {
  const int m = lane & 15;
  const int s = (lane & 16) ? 16 : 0;
  const bf16* p = N + (row0 + m) * DD + kb + s;
  return *(const v16bf*)(p);
}

// B-operand (32x16 bf16, KxN) from TRANSPOSED LDS copy (T[n*DD+k] = M[k][n]).
// Contiguous k order (same as load_A): lane<16: N=lane, elems=K(kb+0..15); lane>=16: N=lane-16, elems=K(kb+16..31)
__device__ __forceinline__ v16bf load_B(const bf16* T, int col0, int kb, int lane) {
  const int n = lane & 15;
  const int s = (lane & 16) ? 16 : 0;
  return *(const v16bf*)(T + (col0 + n) * DD + kb + s);
}

__device__ __forceinline__ v8f wmma_bf(v16bf a, v16bf b, v8f c) {
  return __builtin_amdgcn_wmma_f32_16x16x32_bf16(false, a, false, b, (short)0, c, false, false);
}
// split product: X*Y ~= Xh*Yh + Xh*Yl + Xl*Yh, fp32 accumulate
__device__ __forceinline__ v8f mm3(v16bf ah, v16bf al, v16bf bh, v16bf bl, v8f c) {
  c = wmma_bf(ah, bh, c);
  c = wmma_bf(ah, bl, c);
  c = wmma_bf(al, bh, c);
  return c;
}

__global__ __launch_bounds__(256, 1) void magnus4_kernel(
    const float* __restrict__ t0, const float* __restrict__ hptr,
    const float* __restrict__ y0, const float* __restrict__ A0,
    const float* __restrict__ W, float* __restrict__ out)
{
  // hi/lo split bf16 matrices, row-major (N) and transposed (T) copies. 256 KB (257 KB with sY0/sYr).
  __shared__ bf16 sA1N[2][DD*DD];   // alpha1 rows; later reused as Taylor term T_k (wave-private stripes)
  __shared__ bf16 sA1T[2][DD*DD];   // alpha1 transposed
  __shared__ bf16 sA2N[2][DD*DD];   // alpha2 rows
  __shared__ bf16 sA2T[2][DD*DD];   // alpha2 transposed; later Omega transposed
  __shared__ float sY0[DD];
  __shared__ float sYr[DD];

  const int b    = blockIdx.x;
  const int tid  = threadIdx.x;
  const int lane = tid & 31;
  const int wv   = tid >> 5;          // wave id = tile-row stripe (8 waves x 16 rows)

  const float h   = hptr[0];
  const float tb  = t0[b];
  const float SQ3 = 1.7320508075688772f;
  const float c1  = 0.5f - SQ3 / 6.0f;
  const float c2  = 0.5f + SQ3 / 6.0f;
  const float t1  = tb + c1 * h;
  const float t2  = tb + c2 * h;
  const float w1  = h * (tb + 0.5f * h);  // alpha1 = h*A0 + w1*W
  const float w2  = h * h;                // alpha2 = w2*W   (= h*sqrt3*(t2-t1)*W)

  const float* A0b = A0 + (size_t)b * DD * DD;
  const float* Wb  = W  + (size_t)b * DD * DD;
  float* outA1 = out + (size_t)BB * DD + (size_t)b * DD * DD;
  float* outA2 = out + (size_t)BB * DD + (size_t)BB * DD * DD + (size_t)b * DD * DD;

  if (tid < DD) sY0[tid] = y0[(size_t)b * DD + tid];

  // ---- Stage 1: stream A0/W once, emit A1/A2 outputs, build alpha1/alpha2 in LDS ----
  #pragma unroll
  for (int i = 0; i < 16; ++i) {
    const int f4 = tid + 256 * i;       // coalesced float4 index
    const int e  = f4 << 2;
    const int r  = e >> 7;
    const int c  = e & (DD - 1);
    f32x4 a0v = ((const f32x4*)A0b)[f4];
    f32x4 wvv = ((const f32x4*)Wb)[f4];
    f32x4 A1v = a0v + t1 * wvv;
    f32x4 A2v = a0v + t2 * wvv;
    ((f32x4*)outA1)[f4] = A1v;
    ((f32x4*)outA2)[f4] = A2v;
    f32x4 al1 = h * a0v + w1 * wvv;
    f32x4 al2 = w2 * wvv;
    bf16x4 h1, l1, h2, l2;
    #pragma unroll
    for (int j = 0; j < 4; ++j) {
      bf16 hh = f2bf(al1[j]); h1[j] = hh; l1[j] = f2bf(al1[j] - bf2f(hh));
      bf16 g  = f2bf(al2[j]); h2[j] = g;  l2[j] = f2bf(al2[j] - bf2f(g));
    }
    *(bf16x4*)&sA1N[0][r * DD + c] = h1;
    *(bf16x4*)&sA1N[1][r * DD + c] = l1;
    *(bf16x4*)&sA2N[0][r * DD + c] = h2;
    *(bf16x4*)&sA2N[1][r * DD + c] = l2;
    #pragma unroll
    for (int j = 0; j < 4; ++j) {
      sA1T[0][(c + j) * DD + r] = h1[j];
      sA1T[1][(c + j) * DD + r] = l1[j];
      sA2T[0][(c + j) * DD + r] = h2[j];
      sA2T[1][(c + j) * DD + r] = l2[j];
    }
  }
  __syncthreads();

  // ---- Stage 2: comm = alpha1@alpha2 - alpha2@alpha1 (wave's 16-row stripe) ----
  const v8f vzero = {0.f, 0.f, 0.f, 0.f, 0.f, 0.f, 0.f, 0.f};
  v8f comm[8];
  {
    v8f acc1[8], acc2[8];
    #pragma unroll
    for (int t = 0; t < 8; ++t) { acc1[t] = vzero; acc2[t] = vzero; }
    #pragma unroll
    for (int kk = 0; kk < 4; ++kk) {
      const int kb = kk * 32;
      v16bf a1h = load_A(sA1N[0], wv * 16, kb, lane);
      v16bf a1l = load_A(sA1N[1], wv * 16, kb, lane);
      v16bf a2h = load_A(sA2N[0], wv * 16, kb, lane);
      v16bf a2l = load_A(sA2N[1], wv * 16, kb, lane);
      #pragma unroll
      for (int tn = 0; tn < 8; ++tn) {
        v16bf b2h = load_B(sA2T[0], tn * 16, kb, lane);
        v16bf b2l = load_B(sA2T[1], tn * 16, kb, lane);
        acc1[tn] = mm3(a1h, a1l, b2h, b2l, acc1[tn]);
        v16bf b1h = load_B(sA1T[0], tn * 16, kb, lane);
        v16bf b1l = load_B(sA1T[1], tn * 16, kb, lane);
        acc2[tn] = mm3(a2h, a2l, b1h, b1l, acc2[tn]);
      }
    }
    #pragma unroll
    for (int t = 0; t < 8; ++t) comm[t] = acc1[t] - acc2[t];
  }
  __syncthreads();   // everyone done reading alpha buffers

  // ---- Stage 3: Omega = alpha1 - comm/12; U = I + Omega (regs, C layout);
  //      T1 = Omega -> sA1N (wave's stripe, row-major); Omega^T -> sA2T ----
  v8f U[8];
  {
    const int nl = lane & 15;
    const int m0 = wv * 16 + ((lane >> 4) << 3);   // C layout: VGPR r ~ row m0+r, col = nl
    #pragma unroll
    for (int tn = 0; tn < 8; ++tn) {
      const int gn = tn * 16 + nl;
      bf16x8 xh = *(const bf16x8*)&sA1T[0][gn * DD + m0];
      bf16x8 xl = *(const bf16x8*)&sA1T[1][gn * DD + m0];
      bf16x8 oh, ol;
      #pragma unroll
      for (int r = 0; r < 8; ++r) {
        float a1v = bf2f(xh[r]) + bf2f(xl[r]);
        float o   = a1v - comm[tn][r] * (1.0f / 12.0f);
        U[tn][r]  = o + (((m0 + r) == gn) ? 1.0f : 0.0f);
        bf16 hh = f2bf(o);
        oh[r] = hh;
        ol[r] = f2bf(o - bf2f(hh));
      }
      *(bf16x8*)&sA2T[0][gn * DD + m0] = oh;    // Omega^T (contiguous)
      *(bf16x8*)&sA2T[1][gn * DD + m0] = ol;
      #pragma unroll
      for (int r = 0; r < 8; ++r) {             // Omega rows = T1 (scattered)
        sA1N[0][(m0 + r) * DD + gn] = oh[r];
        sA1N[1][(m0 + r) * DD + gn] = ol[r];
      }
    }
  }
  __syncthreads();   // Omega^T complete before Taylor matmuls

  // ---- Stage 4: Taylor expm. T_k = T_{k-1} @ Omega / k ; U += T_k, k = 2..13.
  //      Left operand stripe is produced & consumed by the same wave -> no barriers. ----
  for (int k = 2; k <= 13; ++k) {
    v8f tacc[8];
    #pragma unroll
    for (int t = 0; t < 8; ++t) tacc[t] = vzero;
    #pragma unroll
    for (int kk = 0; kk < 4; ++kk) {
      const int kb = kk * 32;
      v16bf ah = load_A(sA1N[0], wv * 16, kb, lane);
      v16bf al = load_A(sA1N[1], wv * 16, kb, lane);
      #pragma unroll
      for (int tn = 0; tn < 8; ++tn) {
        v16bf bh = load_B(sA2T[0], tn * 16, kb, lane);
        v16bf bl = load_B(sA2T[1], tn * 16, kb, lane);
        tacc[tn] = mm3(ah, al, bh, bl, tacc[tn]);
      }
    }
    const float invk = 1.0f / (float)k;
    const int nl = lane & 15;
    const int m0 = wv * 16 + ((lane >> 4) << 3);
    #pragma unroll
    for (int tn = 0; tn < 8; ++tn) {
      #pragma unroll
      for (int r = 0; r < 8; ++r) {
        tacc[tn][r] *= invk;
        U[tn][r] += tacc[tn][r];
      }
    }
    if (k < 13) {   // write T_k back (wave-private stripe; LDS ops are in-order per wave)
      #pragma unroll
      for (int tn = 0; tn < 8; ++tn) {
        const int gn = tn * 16 + nl;
        #pragma unroll
        for (int r = 0; r < 8; ++r) {
          float v = tacc[tn][r];
          bf16 hh = f2bf(v);
          sA1N[0][(m0 + r) * DD + gn] = hh;
          sA1N[1][(m0 + r) * DD + gn] = f2bf(v - bf2f(hh));
        }
      }
    }
  }

  // ---- Stage 5: y = U @ y0 via in-wave butterfly reduction over the 16 columns/lane-group ----
  {
    const int nl = lane & 15;
    const int m0 = wv * 16 + ((lane >> 4) << 3);
    #pragma unroll
    for (int r = 0; r < 8; ++r) {
      float p = 0.f;
      #pragma unroll
      for (int tn = 0; tn < 8; ++tn) p += U[tn][r] * sY0[tn * 16 + nl];
      p += __shfl_xor(p, 1, 32);
      p += __shfl_xor(p, 2, 32);
      p += __shfl_xor(p, 4, 32);
      p += __shfl_xor(p, 8, 32);
      if (nl == 0) sYr[m0 + r] = p;   // each row written by exactly one lane
    }
  }
  __syncthreads();
  if (tid < DD) out[(size_t)b * DD + tid] = sYr[tid];
}

extern "C" void kernel_launch(void* const* d_in, const int* in_sizes, int n_in,
                              void* d_out, int out_size, void* d_ws, size_t ws_size,
                              hipStream_t stream) {
  (void)in_sizes; (void)n_in; (void)d_ws; (void)ws_size; (void)out_size;
  const float* t0 = (const float*)d_in[0];
  const float* h  = (const float*)d_in[1];
  const float* y0 = (const float*)d_in[2];
  const float* A0 = (const float*)d_in[3];
  const float* W  = (const float*)d_in[4];
  magnus4_kernel<<<dim3(BB), dim3(256), 0, stream>>>(t0, h, y0, A0, W, (float*)d_out);
}
